// Attention_36644660969693
// MI455X (gfx1250) — compile-verified, matrix operand order checked by reading, not run on hardware
//
#include <hip/hip_runtime.h>
#include <hip/hip_bf16.h>

// ---------------------------------------------------------------------------
// Types for WMMA fragments (CDNA5 gfx1250, wave32)
// ---------------------------------------------------------------------------
typedef __attribute__((ext_vector_type(16))) __bf16      v16bf;
typedef __attribute__((ext_vector_type(8)))  float       v8f;
typedef __attribute__((ext_vector_type(4)))  unsigned int v4u;

union FragU { v4u q[2]; v16bf v; };

// Round-to-nearest-even f32 -> bf16 (bit-level, no header dependence)
__device__ __forceinline__ unsigned short f2bf(float f) {
    unsigned int u = __builtin_bit_cast(unsigned int, f);
    u = (u + 0x7FFFu + ((u >> 16) & 1u)) >> 16;
    return (unsigned short)u;
}

// Load one 16x32 bf16 A/B fragment from a row-major [16 x stride] tile.
// ISA layout (CDNA5 ISA): lane L holds row L&15; kb = (L<16 ? 0 : 8);
// elements 0..7 = K kb..kb+7, elements 8..15 = K kb+16..kb+23.
__device__ __forceinline__ v16bf ld_frag(const unsigned short* base, int stride) {
    const int lane = threadIdx.x & 31;
    const int r    = lane & 15;
    const int kb   = (lane < 16) ? 0 : 8;
    const unsigned short* p = base + (size_t)r * stride + kb;
    FragU u;
    u.q[0] = *(const v4u*)(p);
    u.q[1] = *(const v4u*)(p + 16);
    return u.v;
}

#define WMMA_BF16(A, B, C) \
    __builtin_amdgcn_wmma_f32_16x16x32_bf16(false, (A), false, (B), (short)0, (C), false, false)

// 16-lane-half reductions (wave32: xor masks <16 keep the two halves separate)
__device__ __forceinline__ float half_max(float v) {
    v = fmaxf(v, __shfl_xor(v, 1, 32));
    v = fmaxf(v, __shfl_xor(v, 2, 32));
    v = fmaxf(v, __shfl_xor(v, 4, 32));
    v = fmaxf(v, __shfl_xor(v, 8, 32));
    return v;
}
__device__ __forceinline__ float half_sum(float v) {
    v += __shfl_xor(v, 1, 32);
    v += __shfl_xor(v, 2, 32);
    v += __shfl_xor(v, 4, 32);
    v += __shfl_xor(v, 8, 32);
    return v;
}

// ---------------------------------------------------------------------------
// Conversions
// ---------------------------------------------------------------------------
__global__ void __launch_bounds__(256) cvt_bf16_kernel(const float* __restrict__ s,
                                                       unsigned short* __restrict__ d, int n) {
    int i = blockIdx.x * 256 + threadIdx.x;
    if (i < n) d[i] = f2bf(s[i]);
}

// dst[c*rows + r] = src[r*cols + c]  (store W^T so B-frags load contiguously)
__global__ void __launch_bounds__(256) cvt_transpose_kernel(const float* __restrict__ s,
                                                            unsigned short* __restrict__ d,
                                                            int rows, int cols) {
    int i = blockIdx.x * 256 + threadIdx.x;
    if (i < rows * cols) {
        int r = i / cols, c = i - r * cols;
        d[(size_t)c * rows + r] = f2bf(s[i]);
    }
}

// ---------------------------------------------------------------------------
// QKV projection: [8192,768] x [768,2304] -> scatter into q / k / v^T (bf16)
// One wave = 64x64 output tile (16 WMMA accs, 1 load per WMMA per k-step).
// ---------------------------------------------------------------------------
__global__ void __launch_bounds__(256) qkv_gemm_kernel(
        const unsigned short* __restrict__ xb,    // [8192, 768]
        const unsigned short* __restrict__ wT,    // [2304, 768]
        unsigned short* __restrict__ qout,        // [96, 1024, 64], pre-scaled
        unsigned short* __restrict__ kout,        // [96, 1024, 64]
        unsigned short* __restrict__ vTout) {     // [96, 64, 1024]
    const int lane = threadIdx.x & 31;
    const int wave = blockIdx.x * 8 + (threadIdx.x >> 5);
    const int mt = wave / 36, nt = wave - mt * 36;       // 128 x 36 tiles
    const int m0 = mt * 64, n0 = nt * 64;

    v8f acc[4][4] = {};
    const unsigned short* ap = xb + (size_t)m0 * 768;
    const unsigned short* bp = wT + (size_t)n0 * 768;

    for (int kx = 0; kx < 768; kx += 32) {
        v16bf a[4], b[4];
        #pragma unroll
        for (int i = 0; i < 4; ++i) a[i] = ld_frag(ap + (size_t)(16 * i) * 768 + kx, 768);
        #pragma unroll
        for (int i = 0; i < 4; ++i) b[i] = ld_frag(bp + (size_t)(16 * i) * 768 + kx, 768);
        #pragma unroll
        for (int mi = 0; mi < 4; ++mi)
        #pragma unroll
        for (int ni = 0; ni < 4; ++ni)
            acc[mi][ni] = WMMA_BF16(a[mi], b[ni], acc[mi][ni]);
    }

    // C layout: element i -> row i + (lane<16?0:8), col = lane&15
    const int rb = ((lane >> 4) << 3);
    const int cl = lane & 15;
    #pragma unroll
    for (int mi = 0; mi < 4; ++mi)
    #pragma unroll
    for (int ni = 0; ni < 4; ++ni)
    #pragma unroll
    for (int i = 0; i < 8; ++i) {
        int row = m0 + mi * 16 + rb + i;
        int col = n0 + ni * 16 + cl;
        float val = acc[mi][ni][i];
        int which = col / 768, c = col - which * 768;    // 0=q 1=k 2=v
        int head = c >> 6, hd = c & 63;
        int b = row >> 10, tok = row & 1023;
        size_t bh = (size_t)(b * 12 + head);
        if (which == 0)      qout[(bh * 1024 + tok) * 64 + hd] = f2bf(val * 0.125f);
        else if (which == 1) kout[(bh * 1024 + tok) * 64 + hd] = f2bf(val);
        else                 vTout[(bh * 64 + hd) * 1024 + tok] = f2bf(val);
    }
}

// ---------------------------------------------------------------------------
// Flash attention: one wave = 16 query rows of one (b,h). 4 waves / block.
// ---------------------------------------------------------------------------
__global__ void __launch_bounds__(128) attn_kernel(
        const unsigned short* __restrict__ q,     // [96,1024,64] (scaled)
        const unsigned short* __restrict__ kmat,  // [96,1024,64]
        const unsigned short* __restrict__ vT,    // [96,64,1024]
        unsigned short* __restrict__ aout) {      // [8192,768]
    __shared__ __align__(16) unsigned short ptile[4][16 * 32];

    const int lane = threadIdx.x & 31;
    const int w    = threadIdx.x >> 5;
    const int wave = blockIdx.x * 4 + w;
    const int bh = wave >> 6;        // 0..95
    const int qt = wave & 63;        // query tile

    const unsigned short* qb = q + ((size_t)bh * 1024 + qt * 16) * 64;
    v16bf aq0 = ld_frag(qb, 64);        // K = hd 0..31
    v16bf aq1 = ld_frag(qb + 32, 64);   // K = hd 32..63

    v8f o0 = {}, o1 = {}, o2 = {}, o3 = {};
    float mrow[8], lrow[8];
    #pragma unroll
    for (int i = 0; i < 8; ++i) { mrow[i] = -1e30f; lrow[i] = 0.0f; }

    const unsigned short* kbase = kmat + (size_t)bh * 1024 * 64;
    const unsigned short* vbase = vT + (size_t)bh * 64 * 1024;
    unsigned short* myp = ptile[w];
    const int rb = ((lane >> 4) << 3);
    const int cl = lane & 15;

    for (int kt = 0; kt < 1024; kt += 32) {
        // S = Q (16x64) @ K^T  -> two 16x16 f32 frags (keys kt.. / kt+16..)
        v16bf bk00 = ld_frag(kbase + (size_t)kt * 64, 64);
        v16bf bk01 = ld_frag(kbase + (size_t)kt * 64 + 32, 64);
        v16bf bk10 = ld_frag(kbase + (size_t)(kt + 16) * 64, 64);
        v16bf bk11 = ld_frag(kbase + (size_t)(kt + 16) * 64 + 32, 64);
        v8f z = {};
        v8f s0 = WMMA_BF16(aq0, bk00, z); s0 = WMMA_BF16(aq1, bk01, s0);
        v8f s1 = WMMA_BF16(aq0, bk10, z); s1 = WMMA_BF16(aq1, bk11, s1);

        // Online softmax (row stats across the 16-lane half)
        float corr[8];
        #pragma unroll
        for (int i = 0; i < 8; ++i) {
            float rm = half_max(fmaxf(s0[i], s1[i]));
            float nm = fmaxf(mrow[i], rm);
            corr[i]  = __expf(mrow[i] - nm);
            mrow[i]  = nm;
            float p0 = __expf(s0[i] - nm);
            float p1 = __expf(s1[i] - nm);
            s0[i] = p0; s1[i] = p1;
            lrow[i] = lrow[i] * corr[i] + half_sum(p0 + p1);
            o0[i] *= corr[i]; o1[i] *= corr[i]; o2[i] *= corr[i]; o3[i] *= corr[i];
        }

        // P (C layout) -> LDS -> A-fragment layout (cross-lane transpose)
        asm volatile("" ::: "memory");
        #pragma unroll
        for (int i = 0; i < 8; ++i) {
            myp[(rb + i) * 32 + cl]      = f2bf(s0[i]);
            myp[(rb + i) * 32 + 16 + cl] = f2bf(s1[i]);
        }
        asm volatile("s_wait_dscnt 0x0" ::: "memory");   // DS in-order per wave
        v16bf ap = ld_frag(myp, 32);                     // 16x32 P fragment
        asm volatile("" ::: "memory");

        // O += P (16x32) @ V (32 x 64); V^T rows are contiguous in keys
        o0 = WMMA_BF16(ap, ld_frag(vbase + (size_t)0  * 1024 + kt, 1024), o0);
        o1 = WMMA_BF16(ap, ld_frag(vbase + (size_t)16 * 1024 + kt, 1024), o1);
        o2 = WMMA_BF16(ap, ld_frag(vbase + (size_t)32 * 1024 + kt, 1024), o2);
        o3 = WMMA_BF16(ap, ld_frag(vbase + (size_t)48 * 1024 + kt, 1024), o3);
    }

    // Normalize and store to [b, tok, head*64+hd] bf16
    float inv[8];
    #pragma unroll
    for (int i = 0; i < 8; ++i) inv[i] = 1.0f / lrow[i];
    const int b = bh / 12, h = bh - b * 12;
    #pragma unroll
    for (int i = 0; i < 8; ++i) {
        int tok = qt * 16 + rb + i;
        size_t rowoff = ((size_t)b * 1024 + tok) * 768 + h * 64 + cl;
        aout[rowoff +  0] = f2bf(o0[i] * inv[i]);
        aout[rowoff + 16] = f2bf(o1[i] * inv[i]);
        aout[rowoff + 32] = f2bf(o2[i] * inv[i]);
        aout[rowoff + 48] = f2bf(o3[i] * inv[i]);
    }
}

// ---------------------------------------------------------------------------
// Output projection: [8192,768] x [768,768] -> fp32 d_out
// One wave = 64x64 output tile.
// ---------------------------------------------------------------------------
__global__ void __launch_bounds__(256) oproj_gemm_kernel(
        const unsigned short* __restrict__ ab,    // [8192, 768]
        const unsigned short* __restrict__ wT,    // [768, 768]
        float* __restrict__ out) {
    const int lane = threadIdx.x & 31;
    const int wave = blockIdx.x * 8 + (threadIdx.x >> 5);
    const int mt = wave / 12, nt = wave - mt * 12;       // 128 x 12 tiles
    const int m0 = mt * 64, n0 = nt * 64;

    v8f acc[4][4] = {};
    const unsigned short* ap = ab + (size_t)m0 * 768;
    const unsigned short* bp = wT + (size_t)n0 * 768;

    for (int kx = 0; kx < 768; kx += 32) {
        v16bf a[4], b[4];
        #pragma unroll
        for (int i = 0; i < 4; ++i) a[i] = ld_frag(ap + (size_t)(16 * i) * 768 + kx, 768);
        #pragma unroll
        for (int i = 0; i < 4; ++i) b[i] = ld_frag(bp + (size_t)(16 * i) * 768 + kx, 768);
        #pragma unroll
        for (int mi = 0; mi < 4; ++mi)
        #pragma unroll
        for (int ni = 0; ni < 4; ++ni)
            acc[mi][ni] = WMMA_BF16(a[mi], b[ni], acc[mi][ni]);
    }

    const int rb = ((lane >> 4) << 3);
    const int cl = lane & 15;
    #pragma unroll
    for (int mi = 0; mi < 4; ++mi)
    #pragma unroll
    for (int ni = 0; ni < 4; ++ni)
    #pragma unroll
    for (int i = 0; i < 8; ++i) {
        int row = m0 + mi * 16 + rb + i;
        int col = n0 + ni * 16 + cl;
        out[(size_t)row * 768 + col] = acc[mi][ni][i];
    }
}

// ---------------------------------------------------------------------------
// Host launcher
// ---------------------------------------------------------------------------
extern "C" void kernel_launch(void* const* d_in, const int* in_sizes, int n_in,
                              void* d_out, int out_size, void* d_ws, size_t ws_size,
                              hipStream_t stream) {
    (void)in_sizes; (void)n_in; (void)out_size; (void)ws_size;
    const float* x     = (const float*)d_in[0];   // [8,32,32,768]
    const float* w_qkv = (const float*)d_in[1];   // [768,2304]
    const float* w_o   = (const float*)d_in[2];   // [768,768]
    float* out = (float*)d_out;                   // [8,32,32,768]

    char* ws = (char*)d_ws;
    size_t off = 0;
    auto take = [&](size_t bytes) { void* p = ws + off; off += (bytes + 255) & ~(size_t)255; return p; };

    unsigned short* xb    = (unsigned short*)take((size_t)8192 * 768 * 2);   // also reused as attn_out
    unsigned short* wqkvT = (unsigned short*)take((size_t)2304 * 768 * 2);
    unsigned short* woT   = (unsigned short*)take((size_t)768 * 768 * 2);
    unsigned short* qbuf  = (unsigned short*)take((size_t)96 * 1024 * 64 * 2);
    unsigned short* kbuf  = (unsigned short*)take((size_t)96 * 1024 * 64 * 2);
    unsigned short* vTbuf = (unsigned short*)take((size_t)96 * 64 * 1024 * 2);
    unsigned short* attn_out = xb;  // x is dead after the QKV GEMM

    // 1. Precision conversion / weight transposes
    cvt_bf16_kernel<<<(8192 * 768 + 255) / 256, 256, 0, stream>>>(x, xb, 8192 * 768);
    cvt_transpose_kernel<<<(768 * 2304 + 255) / 256, 256, 0, stream>>>(w_qkv, wqkvT, 768, 2304);
    cvt_transpose_kernel<<<(768 * 768 + 255) / 256, 256, 0, stream>>>(w_o, woT, 768, 768);

    // 2. QKV projection (128 M-tiles x 36 N-tiles = 4608 waves, 8 per block)
    qkv_gemm_kernel<<<576, 256, 0, stream>>>(xb, wqkvT, qbuf, kbuf, vTbuf);

    // 3. Flash attention (96 heads x 64 query tiles = 6144 waves, 4 per block)
    attn_kernel<<<1536, 128, 0, stream>>>(qbuf, kbuf, vTbuf, attn_out);

    // 4. Output projection (128 x 12 tiles = 1536 waves, 8 per block)
    oproj_gemm_kernel<<<192, 256, 0, stream>>>(attn_out, woT, out);
}
